// BinaryConv2d_Fusion_Decrease_58025008169025
// MI455X (gfx1250) — matrix operand order repaired and checked by reading, not compiled, not run on hardware
//
#include <hip/hip_runtime.h>
#include <hip/hip_bf16.h>
#include <stdint.h>

typedef __attribute__((ext_vector_type(16))) __bf16       v16bf;
typedef __attribute__((ext_vector_type(8)))  __bf16       v8bf;
typedef __attribute__((ext_vector_type(8)))  float        v8f;
typedef __attribute__((ext_vector_type(4)))  unsigned int u32x4;
typedef __attribute__((ext_vector_type(8)))  int          i32x8;
typedef __attribute__((ext_vector_type(4)))  int          i32x4;

#define CIN    512
#define COUT   256
#define HW     16384
#define NBATCH 16
#define NPIX   (NBATCH * HW)      // 262144 pixels
#define BNT    128                // pixel tile per workgroup
#define BK     32                 // K (channels) per step
#define LDP    129                // padded LDS pitch in dwords (128 px + 1 pad)
#define XTF    (BK * LDP)         // floats per LDS buffer (4128)
#define KSTEPS (CIN / BK)         // 16

#ifndef __has_builtin
#define __has_builtin(x) 0
#endif

#if __has_builtin(__builtin_amdgcn_tensor_load_to_lds) && __has_builtin(__builtin_amdgcn_s_wait_tensorcnt)
#define HAVE_TDM 1
#else
#define HAVE_TDM 0
#endif

#if HAVE_TDM
// Issue a TDM load of a [BK channels x BNT pixels] f32 tile from global to LDS.
// Tile walk: dim0 = 128 contiguous pixels (512B rows), dim1 = 32 channels with
// stride HW elements (64KB). LDS is written contiguously with 1 dword of pad
// every 128 dwords (pad_interval code 6 = 128 dwords, pad_amount code 0 = 1),
// giving the 129-dword pitch used by the compute waves.
__device__ __forceinline__ void tdm_load_tile(uint64_t gaddr, unsigned int lds_off) {
  u32x4 g0;
  g0[0] = 1u;                                       // count=1, user descriptor
  g0[1] = lds_off;                                  // LDS byte offset
  g0[2] = (unsigned int)gaddr;                      // global addr [31:0]
  g0[3] = (unsigned int)(gaddr >> 32) | (2u << 30); // addr [56:32] | type=2
  i32x8 g1;
  g1[0] = (2 << 16) | (1 << 20) | (6 << 22);        // data_size=4B, pad_en, interval=128dw, amount=1dw
  g1[1] = (int)(16384u << 16);                      // tensor_dim0 = 16384 (lo16 at bits 63:48)
  g1[2] = (int)(512u << 16);                        // dim0 hi16 = 0 | tensor_dim1 = 512 lo16
  g1[3] = (int)(128u << 16);                        // dim1 hi16 = 0 | tile_dim0 = 128
  g1[4] = 32;                                       // tile_dim1 = 32, tile_dim2 = 0
  g1[5] = 16384;                                    // tensor_dim0_stride lo32 (elements)
  g1[6] = 0;
  g1[7] = 0;
  i32x4 z4 = {0, 0, 0, 0};
#if defined(__clang_major__) && (__clang_major__ >= 23)
  i32x8 z8 = {0, 0, 0, 0, 0, 0, 0, 0};
  __builtin_amdgcn_tensor_load_to_lds(g0, g1, z4, z4, z8, 0);
#else
  __builtin_amdgcn_tensor_load_to_lds(g0, g1, z4, z4, 0);
#endif
  asm volatile("" ::: "memory");
}
#endif

// ---------------------------------------------------------------------------
// Kernel 1: binarize weights to bf16 (+/-1) and zero the stats accumulators.
// ---------------------------------------------------------------------------
__global__ __launch_bounds__(256) void prep_kernel(const float* __restrict__ w,
                                                   unsigned short* __restrict__ wbu,
                                                   float* __restrict__ stats) {
  const int idx = blockIdx.x * 256 + threadIdx.x;
  if (idx < COUT * CIN) {
    const float v = w[idx];
    ((__bf16*)wbu)[idx] = (__bf16)((v < 0.0f) ? -1.0f : 1.0f);
  }
  if (idx < 2 * COUT) stats[idx] = 0.0f;
}

// ---------------------------------------------------------------------------
// Kernel 2: GEMM out[b,o,q] = sum_c Wb[o,c] * x[b,c,q] via bf16 WMMA.
// Block tile: 128(M) x 128(N). 8 waves (2 M x 4 N), 64x32 per wave.
// ---------------------------------------------------------------------------
__global__ __launch_bounds__(256) void gemm_kernel(const float* __restrict__ x,
                                                   const unsigned short* __restrict__ wbu,
                                                   float* __restrict__ out) {
  __shared__ float xt[2][XTF];

  const __bf16* wb = (const __bf16*)wbu;
  const int tid   = threadIdx.x;
  const int lane  = tid & 31;
  const int wave  = tid >> 5;
  const int waveM = wave >> 2;              // 0..1
  const int waveN = wave & 3;               // 0..3

  const int p0   = blockIdx.x * BNT;        // global pixel base of this tile
  const int bimg = p0 >> 14;                // image index (HW = 16384)
  const int q0   = p0 & (HW - 1);           // pixel offset within image
  const int m0   = blockIdx.y * 128;        // channel-out base
  const int wm0  = m0 + waveM * 64;
  const int ln0  = waveN * 32;              // local pixel base of this wave

  const float* xbase = x + ((size_t)bimg * CIN) * HW + q0;

  v8f zero = {};
  v8f acc[4][2];
#pragma unroll
  for (int i = 0; i < 4; ++i)
#pragma unroll
    for (int j = 0; j < 2; ++j) acc[i][j] = zero;

  const int lh      = lane >> 4;            // lane half: 0 / 1
  const int ll      = lane & 15;
  const int khalf8  = lh * 8;               // A-matrix K half (ISA 16-bit A layout)
  const int khalf16 = lh * 16;              // unused; B fragments use khalf8 (same K order as A)

  // Per-lane base of its A rows: row m = wm0 + i*16 + ll, K window moves with c0.
  const __bf16* arow = wb + (size_t)(wm0 + ll) * CIN + khalf8;

#if HAVE_TDM
  const unsigned int lds0  = (unsigned int)(uintptr_t)(&xt[0][0]);
  const uint64_t     gbase = (uint64_t)(uintptr_t)xbase;
  if (wave == 0) tdm_load_tile(gbase, lds0);  // prologue: stage K-step 0 into buf 0
#endif

#pragma unroll 1
  for (int kb = 0; kb < KSTEPS; ++kb) {
    const int c0 = kb * BK;

#if HAVE_TDM
    if (wave == 0) __builtin_amdgcn_s_wait_tensorcnt(0);
    __syncthreads();  // buf[kb&1] ready; everyone done with buf[(kb+1)&1]
    if (wave == 0 && (kb + 1) < KSTEPS)
      tdm_load_tile(gbase + (uint64_t)(c0 + BK) * HW * 4u,
                    lds0 + (unsigned int)(((kb + 1) & 1) * (XTF * 4)));
    const float* xs = &xt[kb & 1][0];
#else
    __syncthreads();
    {
      const int pp  = tid & 127;
      const int kk0 = (tid >> 7) * 16;
#pragma unroll
      for (int i = 0; i < 16; ++i)
        xt[0][(kk0 + i) * LDP + pp] = xbase[(size_t)(c0 + kk0 + i) * HW + pp];
    }
    __syncthreads();
    const float* xs = &xt[0][0];
#endif

    // Preload ALL A fragments first (8 independent b128 loads -> one clause,
    // one loadcnt wait ahead of the WMMA burst instead of four).
    v16bf afr[4];
#pragma unroll
    for (int i = 0; i < 4; ++i) {
      const v8bf* ap = (const v8bf*)(arow + (size_t)i * 16 * CIN + c0);
      const v8bf  lo = ap[0];   // K = khalf8 .. +7
      const v8bf  hi = ap[2];   // K = 16+khalf8 .. +7 (16 bf16 = 2*v8bf ahead)
      afr[i] = __builtin_shufflevector(lo, hi, 0, 1, 2, 3, 4, 5, 6, 7,
                                       8, 9, 10, 11, 12, 13, 14, 15);
    }

    // Build B fragments (32K x 16N): lane ll holds pixel column, K order matches A (khalf8 + (t&7) + 16*(t>>3)).
    v16bf bfr[2];
#pragma unroll
    for (int j = 0; j < 2; ++j) {
      const int pp = ln0 + j * 16 + ll;
#pragma unroll
      for (int t = 0; t < 16; ++t)
        bfr[j][t] = (__bf16)xs[(khalf8 + (t & 7) + ((t >> 3) << 4)) * LDP + pp];
    }

    // 8 back-to-back WMMAs (all operands resident).
#pragma unroll
    for (int i = 0; i < 4; ++i) {
      acc[i][0] = __builtin_amdgcn_wmma_f32_16x16x32_bf16(
          false, afr[i], false, bfr[0], (short)0, acc[i][0], false, false);
      acc[i][1] = __builtin_amdgcn_wmma_f32_16x16x32_bf16(
          false, afr[i], false, bfr[1], (short)0, acc[i][1], false, false);
    }
  }

  // Epilogue: store raw conv output. C/D layout: VGPR r -> M = r + 8*laneHalf, N = lane&15.
  float* obase = out + ((size_t)bimg * COUT) * HW + q0;
#pragma unroll
  for (int i = 0; i < 4; ++i) {
#pragma unroll
    for (int j = 0; j < 2; ++j) {
      const int q    = ln0 + j * 16 + ll;
      const int mrow = wm0 + i * 16 + lh * 8;
#pragma unroll
      for (int r = 0; r < 8; ++r)
        obase[(size_t)(mrow + r) * HW + q] = acc[i][j][r];
    }
  }
}

// ---------------------------------------------------------------------------
// Kernel 3: per-channel sum / sum-of-squares over (B,H,W).
// One block per (b,o) plane of 16384 elements.
// ---------------------------------------------------------------------------
__global__ __launch_bounds__(256) void stats_kernel(const float* __restrict__ out,
                                                    float* __restrict__ ssum,
                                                    float* __restrict__ ssq) {
  const int plane = blockIdx.x;       // b*COUT + o
  const int o     = plane & (COUT - 1);
  const float4* p = (const float4*)(out + (size_t)plane * HW);
  float s = 0.0f, q = 0.0f;
  for (int i = threadIdx.x; i < HW / 4; i += 256) {
    const float4 v = p[i];
    s += v.x + v.y + v.z + v.w;
    q += v.x * v.x + v.y * v.y + v.z * v.z + v.w * v.w;
  }
  __shared__ float sh[2][256];
  sh[0][threadIdx.x] = s;
  sh[1][threadIdx.x] = q;
  __syncthreads();
  for (int ofs = 128; ofs > 0; ofs >>= 1) {
    if (threadIdx.x < ofs) {
      sh[0][threadIdx.x] += sh[0][threadIdx.x + ofs];
      sh[1][threadIdx.x] += sh[1][threadIdx.x + ofs];
    }
    __syncthreads();
  }
  if (threadIdx.x == 0) {
    atomicAdd(&ssum[o], sh[0][0]);
    atomicAdd(&ssq[o], sh[1][0]);
  }
}

// ---------------------------------------------------------------------------
// Kernel 4: BatchNorm (batch stats, biased var) + ReLU, in place.
// ---------------------------------------------------------------------------
__global__ __launch_bounds__(256) void bn_relu_kernel(float* __restrict__ out,
                                                      const float* __restrict__ ssum,
                                                      const float* __restrict__ ssq,
                                                      const float* __restrict__ gamma,
                                                      const float* __restrict__ beta) {
  const int plane = blockIdx.x;
  const int o     = plane & (COUT - 1);
  const float npinv = 1.0f / (float)NPIX;
  const float mean  = ssum[o] * npinv;
  const float var   = ssq[o] * npinv - mean * mean;
  const float inv   = gamma[o] * rsqrtf(var + 1e-5f);
  const float shift = beta[o] - mean * inv;
  float4* p = (float4*)(out + (size_t)plane * HW);
  for (int i = threadIdx.x; i < HW / 4; i += 256) {
    float4 v = p[i];
    v.x = fmaxf(fmaf(v.x, inv, shift), 0.0f);
    v.y = fmaxf(fmaf(v.y, inv, shift), 0.0f);
    v.z = fmaxf(fmaf(v.z, inv, shift), 0.0f);
    v.w = fmaxf(fmaf(v.w, inv, shift), 0.0f);
    p[i] = v;
  }
}

// ---------------------------------------------------------------------------
extern "C" void kernel_launch(void* const* d_in, const int* in_sizes, int n_in,
                              void* d_out, int out_size, void* d_ws, size_t ws_size,
                              hipStream_t stream) {
  const float* x      = (const float*)d_in[0];  // [16,512,128,128]
  const float* weight = (const float*)d_in[1];  // [256,512]
  const float* gamma  = (const float*)d_in[2];  // [256]
  const float* beta   = (const float*)d_in[3];  // [256]
  float* out = (float*)d_out;                   // [16,256,128,128]

  char* ws = (char*)d_ws;
  unsigned short* wb = (unsigned short*)ws;               // 256*512*2 = 262144 B
  float* stats = (float*)(ws + (size_t)COUT * CIN * 2);   // sum[256] ++ sumsq[256]

  prep_kernel<<<(COUT * CIN) / 256, 256, 0, stream>>>(weight, wb, stats);

  dim3 grid(NPIX / BNT, COUT / 128, 1);                   // 2048 x 2
  gemm_kernel<<<grid, 256, 0, stream>>>(x, wb, out);

  stats_kernel<<<NBATCH * COUT, 256, 0, stream>>>(out, stats, stats + COUT);
  bn_relu_kernel<<<NBATCH * COUT, 256, 0, stream>>>(out, stats, stats + COUT, gamma, beta);
}
